// TestAGMMModel_3032246911521
// MI455X (gfx1250) — matrix operand order repaired and checked by reading, not compiled, not run on hardware
//
#include <hip/hip_runtime.h>

typedef __attribute__((ext_vector_type(16))) __bf16          v16bf;
typedef __attribute__((ext_vector_type(8)))  float           v8f;
typedef __attribute__((ext_vector_type(4)))  float           f32x4;

constexpr int KD = 2048;   // hidden (K)
constexpr int ND = 2048;   // output features (N) = rows of W

// Split f = hi + lo (both bf16, RNE via native cvt).
// hi = bf16(f); lo = bf16(f - hi).  f - hi is exact in f32, and widening
// bf16 -> f32 is exact (shift), so |f - (hi+lo)| <~ 2^-17 |f|.
__device__ __forceinline__ void split_bf(float f, __bf16& h, __bf16& l) {
  h = (__bf16)f;              // v_cvt bf16 (RNE)
  float hf = (float)h;        // exact widen
  l = (__bf16)(f - hf);       // v_cvt bf16 (RNE)
}

__global__ __launch_bounds__(256) void agmm_wmma_bf16x3(
    const float* __restrict__ A,   // [M, K] row-major
    const float* __restrict__ W,   // [N, K] row-major
    float* __restrict__ C)         // [M, N] row-major
{
  // B-fragment-ordered staging of the 128x32 W tile, split hi/lo.
  // Chunk index = n_local*2 + g ; chunk = 16 contiguous bf16 (K = kBase + 16g + e).
  __shared__ v16bf sWh[256];
  __shared__ v16bf sWl[256];

  const int tid  = threadIdx.x;
  const int wave = tid >> 5;
  const int lane = tid & 31;
  const int r    = lane & 15;   // row within 16 (A: M, B: N, D: N)
  const int g    = lane >> 4;   // lane half

  const int nBase = blockIdx.x << 7;   // 128-wide N tile
  const int mBase = blockIdx.y << 7;   // 128-tall M tile

  // Staging: thread tid handles W row n = tid>>1, K half = tid&1 (16 floats).
  const float* wRow = W + (size_t)(nBase + (tid >> 1)) * KD + ((tid & 1) << 4);
  // A fragment: lane's row = mBase + 16*wave + r ; K runs at 16g..16g+15.
  const float* aRow = A + (size_t)(mBase + (wave << 4) + r) * KD + (g << 4);

  v8f acc[8] = {};

  for (int k = 0; k < KD; k += 32) {
    // ---- cooperative W staging: load 16 contiguous f32, split to bf16 hi/lo ----
    f32x4 q0 = *(const f32x4*)(wRow + k);
    f32x4 q1 = *(const f32x4*)(wRow + k + 4);
    f32x4 q2 = *(const f32x4*)(wRow + k + 8);
    f32x4 q3 = *(const f32x4*)(wRow + k + 12);
    float wf[16];
    #pragma unroll
    for (int e = 0; e < 4; ++e) { wf[e] = q0[e]; wf[4+e] = q1[e]; wf[8+e] = q2[e]; wf[12+e] = q3[e]; }
    v16bf wh, wl;
    #pragma unroll
    for (int e = 0; e < 16; ++e) { __bf16 h, l; split_bf(wf[e], h, l); wh[e] = h; wl[e] = l; }
    sWh[tid] = wh;
    sWl[tid] = wl;

    // ---- per-wave A fragment: 16x32 tile, split hi/lo ----
    // A k order used here (same as B staging): K = 16g+e
    f32x4 p0 = *(const f32x4*)(aRow + k);
    f32x4 p1 = *(const f32x4*)(aRow + k + 4);
    f32x4 p2 = *(const f32x4*)(aRow + k + 8);
    f32x4 p3 = *(const f32x4*)(aRow + k + 12);
    float af[16];
    #pragma unroll
    for (int e = 0; e < 4; ++e) { af[e] = p0[e]; af[4+e] = p1[e]; af[8+e] = p2[e]; af[12+e] = p3[e]; }
    v16bf ah, al;
    #pragma unroll
    for (int e = 0; e < 16; ++e) { __bf16 h, l; split_bf(af[e], h, l); ah[e] = h; al[e] = l; }

    __syncthreads();   // W tile visible

    #pragma unroll
    for (int t = 0; t < 8; ++t) {
      const int bi = (((t << 4) + r) << 1) | g;   // B fragment: lane's N = 16t + r
      v16bf bh = sWh[bi];
      v16bf bl = sWl[bi];
      // C += Ahi*Bhi + Alo*Bhi + Ahi*Blo   (lo*lo dropped: ~2^-18 rel)
      acc[t] = __builtin_amdgcn_wmma_f32_16x16x32_bf16(false, ah, false, bh, (short)0, acc[t], false, false);
      acc[t] = __builtin_amdgcn_wmma_f32_16x16x32_bf16(false, al, false, bh, (short)0, acc[t], false, false);
      acc[t] = __builtin_amdgcn_wmma_f32_16x16x32_bf16(false, ah, false, bl, (short)0, acc[t], false, false);
    }

    __syncthreads();   // reads done before next iteration's stores
  }

  // D layout: VGPR j, lane half g -> M = j + 8g ; N = r
  float* cPtr = C + (size_t)(mBase + (wave << 4) + (g << 3)) * ND + nBase + r;
  #pragma unroll
  for (int t = 0; t < 8; ++t) {
    #pragma unroll
    for (int j = 0; j < 8; ++j) {
      cPtr[(size_t)j * ND + (t << 4)] = acc[t][j];
    }
  }
}

extern "C" void kernel_launch(void* const* d_in, const int* in_sizes, int n_in,
                              void* d_out, int out_size, void* d_ws, size_t ws_size,
                              hipStream_t stream) {
  const float* A = (const float*)d_in[0];   // hidden_states [8, 4096, 2048] fp32
  const float* W = (const float*)d_in[1];   // weight [2048, 2048] fp32
  float* C = (float*)d_out;                 // [32768, 2048] fp32

  const int mRows = in_sizes[0] / KD;       // 32768
  dim3 grid(ND / 128, mRows / 128);         // N-tiles fastest -> A tile reuse in L2
  agmm_wmma_bf16x3<<<grid, dim3(256), 0, stream>>>(A, W, C);
}
